// SelfAttention2D_22136261443882
// MI455X (gfx1250) — matrix operand order repaired and checked by reading, not compiled, not run on hardware
//
#include <hip/hip_runtime.h>
#include <hip/hip_bf16.h>

#define B_  4
#define N_  4096
#define C_  256
#define D_  32
#define M_  (B_ * N_)   // 16384 total tokens

#if defined(__AMDGCN__) && \
    __has_builtin(__builtin_amdgcn_global_load_async_to_lds_b128) && \
    __has_builtin(__builtin_amdgcn_s_wait_asynccnt)
#define HAVE_ASYNC_LDS 1
#else
#define HAVE_ASYNC_LDS 0
#endif

typedef __attribute__((ext_vector_type(16))) __bf16 v16bf;
typedef __attribute__((ext_vector_type(8)))  __bf16 v8bf;
typedef __attribute__((ext_vector_type(8)))  float  v8f;

union AFrag { v16bf v; v8bf h[2]; __bf16 e[16]; };
union Pack8 { v8bf v; __bf16 e[8]; };

#if HAVE_ASYNC_LDS
// builtin prototype (from hipcc diagnostic): params are v4i pointers in AS1/AS3
typedef __attribute__((__vector_size__(4 * sizeof(int)))) int v4i_t;
typedef __attribute__((address_space(1))) v4i_t* gv4i_p;
typedef __attribute__((address_space(3))) v4i_t* lv4i_p;

// per-lane 16-byte memory->LDS copy, no data VGPRs, tracked by ASYNCcnt
static __device__ __forceinline__ void async_cp16(const void* g, void* l) {
    __builtin_amdgcn_global_load_async_to_lds_b128((gv4i_p)g, (lv4i_p)l, 0, 0);
}
static __device__ __forceinline__ void async_wait0() {
    __builtin_amdgcn_s_wait_asynccnt(0);
}
#endif

static __device__ __forceinline__ v8f wmma_bf16(v16bf a, v16bf b, v8f c) {
    // 8 args: (neg_a, A, neg_b, B, c_mod, C, reuse_a, reuse_b)
    return __builtin_amdgcn_wmma_f32_16x16x32_bf16(false, a, false, b, (short)0, c,
                                                   false, false);
}

// lane-XOR via ds_swizzle with immediate pattern (no index VGPRs)
template <int MASK>
static __device__ __forceinline__ float swz_xor(float x) {
    return __int_as_float(
        __builtin_amdgcn_ds_swizzle(__float_as_int(x), 0x1f | (MASK << 10)));
}
static __device__ __forceinline__ float rowmax16(float v) {
    v = fmaxf(v, swz_xor<1>(v));
    v = fmaxf(v, swz_xor<2>(v));
    v = fmaxf(v, swz_xor<4>(v));
    v = fmaxf(v, swz_xor<8>(v));
    return v;
}
static __device__ __forceinline__ float rowsum16(float v) {
    v += swz_xor<1>(v);
    v += swz_xor<2>(v);
    v += swz_xor<4>(v);
    v += swz_xor<8>(v);
    return v;
}

// A operand (16x32 bf16), row m, lane-half hi (ISA 7.12.2)
static __device__ __forceinline__ v16bf load_a_global(const __bf16* __restrict__ base,
                                                      int ld, int m, int k0, int hi) {
    AFrag f;
    f.h[0] = *(const v8bf*)(base + (size_t)m * ld + k0 + 8 * hi);
    f.h[1] = *(const v8bf*)(base + (size_t)m * ld + k0 + 8 * hi + 16);
    return f.v;
}

// B operand from transposed LDS tile T[col][40]: two contiguous 16-B loads
static __device__ __forceinline__ v16bf load_b_ldsT(const __bf16* rowT, int hi) {
    AFrag f;
    f.h[0] = *(const v8bf*)(rowT + 8 * hi);
    f.h[1] = *(const v8bf*)(rowT + 8 * hi + 16);
    return f.v;
}

__global__ void cvt_f32_bf16(const float* __restrict__ s, __bf16* __restrict__ d, int n) {
    int i = blockIdx.x * blockDim.x + threadIdx.x;
    int stride = gridDim.x * blockDim.x;
    for (; i < n; i += stride) d[i] = (__bf16)s[i];
}

// WT[col][k] = (bf16) W[k][col]  -- one-time weight transpose (K = 256 rows)
__global__ void cvt_transpose(const float* __restrict__ W, __bf16* __restrict__ WT,
                              int ncol) {
    int i = blockIdx.x * blockDim.x + threadIdx.x;
    int total = 256 * ncol;
    int stride = gridDim.x * blockDim.x;
    for (; i < total; i += stride) {
        int k = i / ncol, c = i % ncol;
        WT[(size_t)c * 256 + k] = (__bf16)W[i];
    }
}

// C[M, NCOL] = A[M,256] @ W[256,NCOL] + bias.
//   TOUT: write bf16 transposed per batch: out[(b*C_+col)*N_ + n]  (for V)
//   RES : out = xres + gamma * (A@W + bias)  (f32, final projection)
template <int NCOL, bool RES, bool TOUT>
__global__ __launch_bounds__(128, 1)
void gemm_wmma(const __bf16* __restrict__ A, const __bf16* __restrict__ WT,
               const float* __restrict__ bias,
               const float* __restrict__ xres, const float* __restrict__ gamma_p,
               __bf16* __restrict__ outb, float* __restrict__ outf) {
    constexpr int NT  = NCOL / 16;
    constexpr int NCH = NCOL / 32;                  // b128 staging chunks per thread
    __shared__ alignas(16) __bf16 Wt[2][NCOL][40];  // W tile transposed: Wt[buf][col][k]

    const int tid  = threadIdx.x;
    const int lane = tid & 31;
    const int wv   = tid >> 5;
    const int ln   = lane & 15;
    const int hi   = lane >> 4;
    const int rowbase = blockIdx.x * 64 + wv * 16;

#if HAVE_ASYNC_LDS
    auto stageW = [&](int kk, int buf) {
#pragma unroll
        for (int i = 0; i < NCH; ++i) {
            int c = tid + 128 * i;
            int col = c >> 2, cc = c & 3;
            async_cp16(&WT[(size_t)col * 256 + kk + cc * 8], &Wt[buf][col][cc * 8]);
        }
    };
#else
    int4 wbuf[NCH];
    auto loadW = [&](int kk) {
#pragma unroll
        for (int i = 0; i < NCH; ++i) {
            int c = tid + 128 * i;
            int col = c >> 2, cc = c & 3;
            wbuf[i] = *(const int4*)&WT[(size_t)col * 256 + kk + cc * 8];
        }
    };
    auto storeW = [&](int buf) {
#pragma unroll
        for (int i = 0; i < NCH; ++i) {
            int c = tid + 128 * i;
            int col = c >> 2, cc = c & 3;
            *(int4*)&Wt[buf][col][cc * 8] = wbuf[i];
        }
    };
#endif

    v8f acc[NT];
#pragma unroll
    for (int t = 0; t < NT; ++t) acc[t] = (v8f){0.f, 0.f, 0.f, 0.f, 0.f, 0.f, 0.f, 0.f};

#if HAVE_ASYNC_LDS
    stageW(0, 0);
    async_wait0();
#else
    loadW(0);
    storeW(0);
#endif
    __syncthreads();
    v16bf af = load_a_global(A, 256, rowbase + ln, 0, hi);

    int cur = 0;
    for (int kk = 0; kk < 256; kk += 32) {
        bool hn = (kk + 32) < 256;
        v16bf afn = af;
        if (hn) {                                   // issue next-tile loads early
#if HAVE_ASYNC_LDS
            stageW(kk + 32, cur ^ 1);
#else
            loadW(kk + 32);
#endif
            afn = load_a_global(A, 256, rowbase + ln, kk + 32, hi);
        }
#pragma unroll
        for (int t = 0; t < NT; ++t)
            acc[t] = wmma_bf16(af, load_b_ldsT(&Wt[cur][t * 16 + ln][0], hi), acc[t]);
        if (hn) {
#if HAVE_ASYNC_LDS
            async_wait0();
#else
            storeW(cur ^ 1);
#endif
        }
        __syncthreads();
        af = afn;
        cur ^= 1;
    }

    float g = 0.f;
    if constexpr (RES) g = *gamma_p;
#pragma unroll
    for (int t = 0; t < NT; ++t) {
        int col = t * 16 + ln;
        if constexpr (TOUT) {
            Pack8 pk;
#pragma unroll
            for (int r = 0; r < 8; ++r) pk.e[r] = (__bf16)(acc[t][r] + bias[col]);
            int row0 = rowbase + 8 * hi;            // 8 row-contiguous results per lane
            int bb = row0 / N_, nn = row0 % N_;
            *(v8bf*)&outb[((size_t)bb * C_ + col) * N_ + nn] = pk.v;
        } else {
#pragma unroll
            for (int r = 0; r < 8; ++r) {
                int row = rowbase + r + 8 * hi;
                float v = acc[t][r] + bias[col];
                if constexpr (RES) {
                    size_t idx = (size_t)row * NCOL + col;
                    outf[idx] = xres[idx] + g * v;
                } else {
                    outb[(size_t)row * NCOL + col] = (__bf16)v;
                }
            }
        }
    }
}

// Flash attention: per (batch, 64-query block), 4 waves x 16 query rows each.
// Double-buffered K/V tiles staged via async memory->LDS copies (no data VGPRs).
__global__ __launch_bounds__(128, 1)
void attn_flash(const __bf16* __restrict__ Qb, const __bf16* __restrict__ Kb,
                const __bf16* __restrict__ VbT, __bf16* __restrict__ Ob) {
    __shared__ alignas(16) __bf16 Kt[2][32][40];     // K tile row-major (B col == K row)
    __shared__ alignas(16) __bf16 Vt[2][256][40];    // V tile transposed: Vt[buf][col][k]
    __shared__ alignas(16) __bf16 Pt[4][16][40];     // per-wave P relayout buffer

    const int tid  = threadIdx.x;
    const int lane = tid & 31;
    const int wv   = tid >> 5;
    const int ln   = lane & 15;
    const int hi   = lane >> 4;
    const int b    = blockIdx.y;
    const int qbase = blockIdx.x * 64 + wv * 16;
    const size_t bn = (size_t)b * N_;
    const float scale = 0.1767766952966369f;         // 1/sqrt(32)

    const __bf16* Kbase = Kb + bn * D_;
    const __bf16* Vbase = VbT + (size_t)b * C_ * N_;

    const int krow = tid >> 2, kcc = tid & 3;        // K tile: 128 b128 chunks

#if HAVE_ASYNC_LDS
    auto stage_tile = [&](int j0, int buf) {
        async_cp16(&Kbase[(size_t)(j0 + krow) * D_ + kcc * 8], &Kt[buf][krow][kcc * 8]);
#pragma unroll
        for (int i = 0; i < 8; ++i) {
            int c = tid + 128 * i;
            int col = c >> 2, cc = c & 3;
            async_cp16(&Vbase[(size_t)col * N_ + j0 + cc * 8], &Vt[buf][col][cc * 8]);
        }
    };
#else
    // register fallback: V staged in two 16-VGPR halves to cap live range
    int4 kbuf;
    int4 vbuf[4];
    auto loadK = [&](int j0) {
        kbuf = *(const int4*)&Kbase[(size_t)(j0 + krow) * D_ + kcc * 8];
    };
    auto storeK = [&](int buf) { *(int4*)&Kt[buf][krow][kcc * 8] = kbuf; };
    auto loadV = [&](int j0, int h) {
#pragma unroll
        for (int i = 0; i < 4; ++i) {
            int c = tid + 128 * (h * 4 + i);
            int col = c >> 2, cc = c & 3;
            vbuf[i] = *(const int4*)&Vbase[(size_t)col * N_ + j0 + cc * 8];
        }
    };
    auto storeV = [&](int buf, int h) {
#pragma unroll
        for (int i = 0; i < 4; ++i) {
            int c = tid + 128 * (h * 4 + i);
            int col = c >> 2, cc = c & 3;
            *(int4*)&Vt[buf][col][cc * 8] = vbuf[i];
        }
    };
#endif

    v16bf qf = load_a_global(Qb + bn * D_, D_, qbase + ln, 0, hi);

    float mrow[8], lrow[8];
    v8f acc[16];
#pragma unroll
    for (int r = 0; r < 8; ++r) { mrow[r] = -1e30f; lrow[r] = 0.f; }
#pragma unroll
    for (int t = 0; t < 16; ++t) acc[t] = (v8f){0.f, 0.f, 0.f, 0.f, 0.f, 0.f, 0.f, 0.f};

#if HAVE_ASYNC_LDS
    stage_tile(0, 0);
    async_wait0();
#else
    loadK(0); loadV(0, 0); storeK(0); storeV(0, 0);
    loadV(0, 1); storeV(0, 1);
#endif
    __syncthreads();

    int cur = 0;
    for (int j0 = 0; j0 < N_; j0 += 32) {
        bool hn = (j0 + 32) < N_;
#if HAVE_ASYNC_LDS
        if (hn) stage_tile(j0 + 32, cur ^ 1);        // mem->LDS flies during compute
#else
        if (hn) { loadK(j0 + 32); loadV(j0 + 32, 0); }
#endif

        // S = Q @ K^T : two 16-key halves; B column n == K row
        v8f s0 = (v8f){0.f, 0.f, 0.f, 0.f, 0.f, 0.f, 0.f, 0.f};
        v8f s1 = s0;
        s0 = wmma_bf16(qf, load_b_ldsT(&Kt[cur][ln][0],      hi), s0);
        s1 = wmma_bf16(qf, load_b_ldsT(&Kt[cur][16 + ln][0], hi), s1);

        // online softmax; row r of the tile lives across one 16-lane half.
#pragma unroll
        for (int r = 0; r < 8; ++r) {
            float e0 = s0[r] * scale, e1 = s1[r] * scale;
            float mx = rowmax16(fmaxf(e0, e1));
            float mnew = fmaxf(mrow[r], mx);
            float corr = __expf(mrow[r] - mnew);
            mrow[r] = mnew;
            float q0 = __expf(e0 - mnew);
            float q1 = __expf(e1 - mnew);
            Pt[wv][r + 8 * hi][ln]      = (__bf16)q0;
            Pt[wv][r + 8 * hi][16 + ln] = (__bf16)q1;
            lrow[r] = lrow[r] * corr + rowsum16(q0 + q1);
#pragma unroll
            for (int t = 0; t < 16; ++t) acc[t][r] *= corr;
        }

#if !HAVE_ASYNC_LDS
        if (hn) { storeK(cur ^ 1); storeV(cur ^ 1, 0); loadV(j0 + 32, 1); }
#endif

        // P relayout read is same-wave (LDS is in-order per wave): no barrier needed
        AFrag pf;
        pf.h[0] = *(const v8bf*)&Pt[wv][ln][8 * hi];
        pf.h[1] = *(const v8bf*)&Pt[wv][ln][8 * hi + 16];

        // O += P(16x32) @ V(32x256) : 16 column tiles from transposed LDS
#pragma unroll
        for (int t = 0; t < 16; ++t)
            acc[t] = wmma_bf16(pf.v, load_b_ldsT(&Vt[cur][t * 16 + ln][0], hi), acc[t]);

        if (hn) {
#if HAVE_ASYNC_LDS
            async_wait0();                           // next buffer fully in LDS
#else
            storeV(cur ^ 1, 1);
#endif
        }
        __syncthreads();
        cur ^= 1;
    }

    // normalize and store O (bf16, row-major for the projection GEMM)
#pragma unroll
    for (int r = 0; r < 8; ++r) {
        float inv = 1.0f / lrow[r];
        int row = qbase + r + 8 * hi;
#pragma unroll
        for (int t = 0; t < 16; ++t)
            Ob[(bn + row) * C_ + t * 16 + ln] = (__bf16)(acc[t][r] * inv);
    }
}

extern "C" void kernel_launch(void* const* d_in, const int* in_sizes, int n_in,
                              void* d_out, int out_size, void* d_ws, size_t ws_size,
                              hipStream_t stream) {
    (void)in_sizes; (void)n_in; (void)out_size; (void)ws_size;
    const float* x     = (const float*)d_in[0];
    const float* Wq    = (const float*)d_in[1];
    const float* bq    = (const float*)d_in[2];
    const float* Wk    = (const float*)d_in[3];
    const float* bk    = (const float*)d_in[4];
    const float* Wv    = (const float*)d_in[5];
    const float* bv    = (const float*)d_in[6];
    const float* Wp    = (const float*)d_in[7];
    const float* bp    = (const float*)d_in[8];
    const float* gamma = (const float*)d_in[9];
    float* out = (float*)d_out;

    char* ws = (char*)d_ws;
    __bf16* xb  = (__bf16*)ws;  ws += (size_t)M_ * C_ * 2;   // 8 MB, row-major
    __bf16* WqT = (__bf16*)ws;  ws += (size_t)C_ * D_ * 2;   // transposed [32][256]
    __bf16* WkT = (__bf16*)ws;  ws += (size_t)C_ * D_ * 2;
    __bf16* WvT = (__bf16*)ws;  ws += (size_t)C_ * C_ * 2;   // [256][256]
    __bf16* WpT = (__bf16*)ws;  ws += (size_t)C_ * C_ * 2;
    __bf16* Qb  = (__bf16*)ws;  ws += (size_t)M_ * D_ * 2;   // 1 MB, row-major
    __bf16* Kb  = (__bf16*)ws;  ws += (size_t)M_ * D_ * 2;   // 1 MB, row-major
    __bf16* VbT = (__bf16*)ws;  ws += (size_t)M_ * C_ * 2;   // 8 MB, [b][col][n]
    __bf16* Ob  = (__bf16*)ws;  ws += (size_t)M_ * C_ * 2;   // 8 MB, row-major

    // 1) f32 -> bf16 conversions (+ one-time weight transposes)
    cvt_f32_bf16 <<<2048, 256, 0, stream>>>(x,  xb,  M_ * C_);
    cvt_transpose<<<32,   256, 0, stream>>>(Wq, WqT, D_);
    cvt_transpose<<<32,   256, 0, stream>>>(Wk, WkT, D_);
    cvt_transpose<<<64,   256, 0, stream>>>(Wv, WvT, C_);
    cvt_transpose<<<64,   256, 0, stream>>>(Wp, WpT, C_);

    // 2) Q/K/V projections (WMMA bf16, f32 accumulate); V written transposed
    gemm_wmma<32,  false, false><<<M_ / 64, 128, 0, stream>>>(xb, WqT, bq, nullptr, nullptr, Qb,  nullptr);
    gemm_wmma<32,  false, false><<<M_ / 64, 128, 0, stream>>>(xb, WkT, bk, nullptr, nullptr, Kb,  nullptr);
    gemm_wmma<256, false, true ><<<M_ / 64, 128, 0, stream>>>(xb, WvT, bv, nullptr, nullptr, VbT, nullptr);

    // 3) flash attention (async double-buffered K/V tiles)
    attn_flash<<<dim3(N_ / 64, B_), 128, 0, stream>>>(Qb, Kb, VbT, Ob);

    // 4) output projection + residual
    gemm_wmma<256, true, false><<<M_ / 64, 128, 0, stream>>>(Ob, WpT, bp, x, gamma, nullptr, out);
}
